// NEP_712964571411
// MI455X (gfx1250) — matrix operand order repaired and checked by reading, not compiled, not run on hardware
//
#include <hip/hip_runtime.h>
#include <hip/hip_bf16.h>

#define N_ATOMS   1048576
#define N_STRUCT  4096
#define ATOMS_PER 256
#define N_RAD     16
#define N_ANG_L   48      // N_ANG * L_MAX
#define N_ELEM    4
#define N_TILES   (N_ATOMS / 16)
#define N_PAIRS   (N_TILES / 2)          // 32 atoms per wave
#define N_BLK     (N_ATOMS / 256)        // 4096 sort blocks
#define WLDS_HALVES 32768                // 64 KB of repacked weights
#define STAGE_HALVES 1024                // 16x64 f16 per wave

typedef __attribute__((ext_vector_type(16))) _Float16 v16h;
typedef __attribute__((ext_vector_type(8)))  _Float16 v8h;
typedef __attribute__((ext_vector_type(8)))  float    v8f;
typedef __attribute__((ext_vector_type(4)))  unsigned int v4u;
typedef __attribute__((ext_vector_type(8)))  int v8i;
typedef __attribute__((ext_vector_type(4)))  int v4i;

#define CAT16(lo, hi) __builtin_shufflevector((lo), (hi), 0,1,2,3,4,5,6,7,8,9,10,11,12,13,14,15)

#if __has_builtin(__builtin_amdgcn_tensor_load_to_lds) && \
    __has_builtin(__builtin_amdgcn_s_wait_tensorcnt)
#define NEP_USE_TDM 1
#else
#define NEP_USE_TDM 0
#endif

// gfx1250 hardware tanh (confirmed lowering to v_tanh_f32); branch-free fallback.
__device__ __forceinline__ float fast_tanh(float x) {
#if __has_builtin(__builtin_amdgcn_tanhf)
  return __builtin_amdgcn_tanhf(x);
#else
  float e = __builtin_amdgcn_exp2f(x * 2.8853900817779268f);   // 2*log2(e)
  return 1.0f - 2.0f * __builtin_amdgcn_rcpf(e + 1.0f);
#endif
}

// ===========================================================================
// Deterministic stable counting sort of atoms by element type.
// ===========================================================================

// (1) per-256-atom-block type histogram (LDS integer atomics: deterministic).
__global__ __launch_bounds__(256) void nep_hist(const int* __restrict__ types,
                                                unsigned* __restrict__ cnt) {
  __shared__ unsigned c[N_ELEM];
  int tl = threadIdx.x;
  if (tl < N_ELEM) c[tl] = 0;
  __syncthreads();
  atomicAdd(&c[types[(size_t)blockIdx.x * 256 + tl]], 1u);
  __syncthreads();
  if (tl < N_ELEM) cnt[blockIdx.x * N_ELEM + tl] = c[tl];
}

// (2) one block: exclusive offsets blockOff[b][t] = typeBase[t] + sum of cnt
// of earlier blocks for type t. 256 threads, 16 blocks each.
__global__ __launch_bounds__(256) void nep_offsets(const unsigned* __restrict__ cnt,
                                                   unsigned* __restrict__ blockOff) {
  __shared__ unsigned s[256];
  __shared__ unsigned typeTot[N_ELEM];
  __shared__ unsigned typeBase[N_ELEM];
  int g = threadIdx.x;
  unsigned local[N_ELEM] = {0, 0, 0, 0};
  for (int i = 0; i < 16; ++i) {
    int b = g * 16 + i;
    local[0] += cnt[b * 4 + 0]; local[1] += cnt[b * 4 + 1];
    local[2] += cnt[b * 4 + 2]; local[3] += cnt[b * 4 + 3];
  }
  unsigned gex[N_ELEM];
  for (int t = 0; t < N_ELEM; ++t) {
    s[g] = local[t];
    __syncthreads();
    for (int off = 1; off < 256; off <<= 1) {       // inclusive Hillis-Steele
      unsigned a = (g >= off) ? s[g - off] : 0u;
      __syncthreads();
      s[g] += a;
      __syncthreads();
    }
    gex[t] = s[g] - local[t];
    if (g == 255) typeTot[t] = s[255];
    __syncthreads();
  }
  if (g == 0) {
    typeBase[0] = 0;
    typeBase[1] = typeTot[0];
    typeBase[2] = typeBase[1] + typeTot[1];
    typeBase[3] = typeBase[2] + typeTot[2];
  }
  __syncthreads();
  for (int t = 0; t < N_ELEM; ++t) {
    unsigned run = typeBase[t] + gex[t];
    for (int i = 0; i < 16; ++i) {
      int b = g * 16 + i;
      blockOff[b * 4 + t] = run;
      run += cnt[b * 4 + t];
    }
  }
}

// (3) stable scatter: rank within block via two packed 16-bit prefix scans.
__global__ __launch_bounds__(256) void nep_scatter(const int* __restrict__ types,
                                                   const unsigned* __restrict__ blockOff,
                                                   int* __restrict__ perm) {
  __shared__ unsigned s0[256], s1[256];
  int tl = threadIdx.x;
  int atom = blockIdx.x * 256 + tl;
  int ty = types[atom];
  unsigned p0 = (ty == 0 ? 1u : 0u) | (ty == 1 ? 0x10000u : 0u);
  unsigned p1 = (ty == 2 ? 1u : 0u) | (ty == 3 ? 0x10000u : 0u);
  s0[tl] = p0; s1[tl] = p1;
  __syncthreads();
  for (int off = 1; off < 256; off <<= 1) {         // inclusive scan, both packed
    unsigned a0 = (tl >= off) ? s0[tl - off] : 0u;
    unsigned a1 = (tl >= off) ? s1[tl - off] : 0u;
    __syncthreads();
    s0[tl] += a0; s1[tl] += a1;
    __syncthreads();
  }
  unsigned inc0 = s0[tl], inc1 = s1[tl];
  unsigned rank = (ty == 0) ? (inc0 & 0xffffu) - 1u
                : (ty == 1) ? (inc0 >> 16) - 1u
                : (ty == 2) ? (inc1 & 0xffffu) - 1u
                :             (inc1 >> 16) - 1u;
  perm[blockOff[blockIdx.x * 4 + ty] + rank] = atom;
}

// ===========================================================================
// Weight repack: W0/W1 [4][64][64] f32 row-major -> per-lane B-fragment f16.
// dst index = ((((l*4+t)*4+n)*2+kb)*32 + lane)*16 + j
// B-fragment element (per CDNA5 ISA 7.12.4 B layout): N = n*16 + lane%16,
// K = kb*32 + (lane>=16 ? 8 : 0) + (j&7) + (j>=8 ? 16 : 0)
// ===========================================================================
__global__ void nep_repack_w(const float* __restrict__ W0,
                             const float* __restrict__ W1,
                             _Float16* __restrict__ dst) {
  int i = blockIdx.x * blockDim.x + threadIdx.x;   // 32768 elements
  int j  = i & 15;
  int L  = (i >> 4)  & 31;
  int kb = (i >> 9)  & 1;
  int n  = (i >> 10) & 3;
  int t  = (i >> 12) & 3;
  int l  = (i >> 14) & 1;
  int K  = kb * 32 + ((L >> 4) << 3) + (j & 7) + ((j >> 3) << 4);
  int N  = (n << 4) + (L & 15);
  const float* W = l ? W1 : W0;
  dst[i] = (_Float16)W[(t * 64 + K) * 64 + N];
}

// ===========================================================================
// Helpers
// ===========================================================================
__device__ __forceinline__ v8h cvt8(float4 a, float4 b) {
  v8h r;
  r[0] = (_Float16)a.x; r[1] = (_Float16)a.y; r[2] = (_Float16)a.z; r[3] = (_Float16)a.w;
  r[4] = (_Float16)b.x; r[5] = (_Float16)b.y; r[6] = (_Float16)b.z; r[7] = (_Float16)b.w;
  return r;
}

__device__ __forceinline__ v16h lds_bfrag(const _Float16* wlds, int l, int t,
                                          int n, int kb, int lane) {
  const _Float16* p = wlds + (size_t)(((((l * 4 + t) * 4 + n) * 2 + kb) * 32 + lane)) * 16;
  return *(const v16h*)p;   // 32 B -> ds_load_b128 pair
}

// Load two X A-fragments (K 0..31 / 32..63) for one atom row from global.
__device__ __forceinline__ void load_x_frags(const float* __restrict__ gr,
                                             const float* __restrict__ ga,
                                             int atom, int hi, v16h& A0, v16h& A1) {
  const float* grp = gr + (size_t)atom * N_RAD;
  const float* gap = ga + (size_t)atom * N_ANG_L;
  float4 r0a = *(const float4*)(grp + hi * 8);
  float4 r0b = *(const float4*)(grp + hi * 8 + 4);
  float4 a0a = *(const float4*)(gap + hi * 8);
  float4 a0b = *(const float4*)(gap + hi * 8 + 4);
  float4 a1a = *(const float4*)(gap + 16 + hi * 8);
  float4 a1b = *(const float4*)(gap + 16 + hi * 8 + 4);
  float4 a2a = *(const float4*)(gap + 32 + hi * 8);
  float4 a2b = *(const float4*)(gap + 32 + hi * 8 + 4);
  A0 = CAT16(cvt8(r0a, r0b), cvt8(a0a, a0b));
  A1 = CAT16(cvt8(a1a, a1b), cvt8(a2a, a2b));
}

// One 64->64 layer for TWO atom tiles sharing the same B fragments.
__device__ __forceinline__ void nep_layer2t(
    const v16h Aa0, const v16h Aa1, const v16h Ab0, const v16h Ab1,
    const _Float16* __restrict__ wlds, int l, int t, int lane, int row,
    const float* __restrict__ bias, v8f* Ca, v8f* Cb) {
#pragma unroll
  for (int n = 0; n < 4; ++n) {
    float bv = bias[t * 64 + n * 16 + row];
    v8f c = {bv, bv, bv, bv, bv, bv, bv, bv};
    v16h B0 = lds_bfrag(wlds, l, t, n, 0, lane);
    v16h B1 = lds_bfrag(wlds, l, t, n, 1, lane);
    Ca[n] = __builtin_amdgcn_wmma_f32_16x16x32_f16(false, Aa0, false, B0, (short)0, c, false, false);
    Ca[n] = __builtin_amdgcn_wmma_f32_16x16x32_f16(false, Aa1, false, B1, (short)0, Ca[n], false, false);
    Cb[n] = __builtin_amdgcn_wmma_f32_16x16x32_f16(false, Ab0, false, B0, (short)0, c, false, false);
    Cb[n] = __builtin_amdgcn_wmma_f32_16x16x32_f16(false, Ab1, false, B1, (short)0, Cb[n], false, false);
  }
}

// tanh + C-layout -> row-major f16 LDS stage (16x64). VGPR r of C holds
// (M = r + 8*hi, N = n*16 + row).
__device__ __forceinline__ void stage_tanh(const v8f* C, _Float16* st,
                                           int row, int hi) {
#pragma unroll
  for (int n = 0; n < 4; ++n)
#pragma unroll
    for (int r = 0; r < 8; ++r)
      st[(r + hi * 8) * 64 + n * 16 + row] = (_Float16)fast_tanh(C[n][r]);
}

// Read A-fragments back from the staged 16x64 tile (A layout).
__device__ __forceinline__ void load_stage_frags(const _Float16* __restrict__ st,
                                                 int row, int hi, v16h& F0, v16h& F1) {
  v8h lo0 = *(const v8h*)(st + row * 64 + hi * 8);
  v8h up0 = *(const v8h*)(st + row * 64 + hi * 8 + 16);
  v8h lo1 = *(const v8h*)(st + row * 64 + 32 + hi * 8);
  v8h up1 = *(const v8h*)(st + row * 64 + 48 + hi * 8);
  F0 = CAT16(lo0, up0);
  F1 = CAT16(lo1, up1);
}

// Final 64->1 dot from the staged tile; half-wave column split + shfl combine.
__device__ __forceinline__ float stage_dot(const _Float16* __restrict__ st,
                                           const float* __restrict__ w2,
                                           int row, int hi) {
  float e = 0.0f;
  const _Float16* hrow = st + row * 64 + hi * 32;
  const float* w = w2 + hi * 32;
#pragma unroll
  for (int q = 0; q < 32; q += 8) {
    v8h hv = *(const v8h*)(hrow + q);
    float4 wa = *(const float4*)(w + q);
    float4 wb = *(const float4*)(w + q + 4);
    e += (float)hv[0] * wa.x + (float)hv[1] * wa.y +
         (float)hv[2] * wa.z + (float)hv[3] * wa.w +
         (float)hv[4] * wb.x + (float)hv[5] * wb.y +
         (float)hv[6] * wb.z + (float)hv[7] * wb.w;
  }
  e += __shfl_xor(e, 16, 32);
  return e;
}

// ===========================================================================
// Main kernel: one wave per 32 SORTED atom positions; 8 waves/block.
// Sorted windows are single-type except at <=3 region boundaries, so the
// type loop runs tmin..tmax (wave-uniform, almost always one iteration).
// LDS: 64 KB weight block (TDM-loaded once per block) + 2 KB stage per wave.
// ===========================================================================
__global__ __launch_bounds__(256) void nep_mlp_wmma(
    const float* __restrict__ gr, const float* __restrict__ ga,
    const int* __restrict__ types, const int* __restrict__ perm,
    const _Float16* __restrict__ wp,
    const float* __restrict__ b0, const float* __restrict__ b1,
    const float* __restrict__ W2, const float* __restrict__ sbias,
    float* __restrict__ e_atom) {
  extern __shared__ __align__(16) char smem[];
  _Float16* wlds = (_Float16*)smem;                                  // 64 KB
  _Float16* st = (_Float16*)(smem + WLDS_HALVES * 2) +
                 (threadIdx.x >> 5) * STAGE_HALVES;                  // 2 KB/wave

  // ---- stage repacked weights into LDS once per block ----
#if NEP_USE_TDM
  if (threadIdx.x < 32) {          // wave 0 issues one TDM descriptor
    // D# group0: count=1 | lds_addr=0 | global_addr | type=2 ("image")
    unsigned long long ga64 = (unsigned long long)(uintptr_t)wp;
    v4u g0;
    g0[0] = 1u;                                        // count=1, user D#
    g0[1] = 0u;                                        // lds_addr = 0
    g0[2] = (unsigned)(ga64 & 0xffffffffu);            // global_addr[31:0]
    g0[3] = (unsigned)((ga64 >> 32) & 0x1ffffffu) | (2u << 30);  // [56:32]|type
    // D# group1: flat 64 KB tile: data_size=8B, tile/tensor dim0=8192, dim1=1
    v8i g1;
    g1[0] = 0x30000;                 // workgroup_mask=0, data_size=3 (8 B)
    g1[1] = (int)(0x2000u << 16);    // tensor_dim0[15:0]=8192  (bits 79:48)
    g1[2] = 0x10000;                 // tensor_dim0 hi=0 | tensor_dim1=1
    g1[3] = (int)(0x2000u << 16);    // tile_dim0 = 8192        (bits 127:112)
    g1[4] = 1;                       // tile_dim1 = 1, tile_dim2 = 0
    g1[5] = 8192;                    // tensor_dim0_stride = 8192
    g1[6] = 0;
    g1[7] = 0;
    v4i z4 = {0, 0, 0, 0};
    v8i z8 = {0, 0, 0, 0, 0, 0, 0, 0};
    __builtin_amdgcn_tensor_load_to_lds(g0, g1, z4, z4, z8, 0);
    __builtin_amdgcn_s_wait_tensorcnt(0);
  }
  __syncthreads();
#else
  {
    const uint4* src = (const uint4*)wp;                 // 4096 x 16 B
    uint4* dst = (uint4*)wlds;
#pragma unroll
    for (int k = 0; k < 16; ++k)
      dst[threadIdx.x + k * 256] = src[threadIdx.x + k * 256];
  }
  __syncthreads();
#endif

  const int lane = threadIdx.x & 31;
  const int wv   = threadIdx.x >> 5;
  const int pair = blockIdx.x * 8 + wv;
  if (pair >= N_PAIRS) return;             // uniform per wave

  const int hi    = lane >> 4;
  const int row   = lane & 15;
  const int posA  = pair * 32 + row;       // sorted positions
  const int atomA = perm[posA];
  const int atomB = perm[posA + 16];

  v16h Aa0, Aa1, Ab0, Ab1;
  load_x_frags(gr, ga, atomA, hi, Aa0, Aa1);
  load_x_frags(gr, ga, atomB, hi, Ab0, Ab1);

  const int   taA = types[atomA];
  const int   taB = types[atomB];
  const float sb  = sbias[0];
  float eA = 0.0f, eB = 0.0f;

  // wave-uniform type range of this sorted window (usually tmin == tmax)
  int tmin = min(taA, taB), tmax = max(taA, taB);
#pragma unroll
  for (int off = 16; off > 0; off >>= 1) {
    tmin = min(tmin, __shfl_xor(tmin, off, 32));
    tmax = max(tmax, __shfl_xor(tmax, off, 32));
  }

  for (int t = tmin; t <= tmax; ++t) {
    v8f Ca[4], Cb[4];
    // layer 0: shared B fragments feed both tiles
    nep_layer2t(Aa0, Aa1, Ab0, Ab1, wlds, 0, t, lane, row, b0, Ca, Cb);
    v16h Ha0, Ha1, Hb0, Hb1;
    stage_tanh(Ca, st, row, hi);               // tile A transpose via LDS
    load_stage_frags(st, row, hi, Ha0, Ha1);
    stage_tanh(Cb, st, row, hi);               // tile B reuses same 2 KB stage
    load_stage_frags(st, row, hi, Hb0, Hb1);
    // layer 1
    nep_layer2t(Ha0, Ha1, Hb0, Hb1, wlds, 1, t, lane, row, b1, Ca, Cb);
    // layer 2: stage each tile's H1 and dot with W2[t]
    const float* w2 = W2 + t * 64;
    stage_tanh(Ca, st, row, hi);
    float ea = stage_dot(st, w2, row, hi) + sb;
    stage_tanh(Cb, st, row, hi);
    float eb = stage_dot(st, w2, row, hi) + sb;
    eA = (taA == t) ? ea : eA;
    eB = (taB == t) ? eb : eB;
  }

  if (hi == 0) {                 // lanes 0..15 hold rows 0..15
    e_atom[atomA] = eA;          // scatter back to original order
    e_atom[atomB] = eB;
  }
}

// ===========================================================================
// Deterministic per-structure reduction: 256 contiguous atoms per structure.
// ===========================================================================
__global__ __launch_bounds__(256) void nep_struct_sum(const float* __restrict__ e_atom,
                                                      float* __restrict__ e_total) {
  __shared__ float s[256];
  int tl = threadIdx.x;
  s[tl] = e_atom[(size_t)blockIdx.x * ATOMS_PER + tl];
  __syncthreads();
#pragma unroll
  for (int off = 128; off > 0; off >>= 1) {
    if (tl < off) s[tl] += s[tl + off];
    __syncthreads();
  }
  if (tl == 0) e_total[blockIdx.x] = s[0];
}

// ===========================================================================
extern "C" void kernel_launch(void* const* d_in, const int* in_sizes, int n_in,
                              void* d_out, int out_size, void* d_ws, size_t ws_size,
                              hipStream_t stream) {
  const float* g_radial  = (const float*)d_in[0];
  const float* g_angular = (const float*)d_in[1];
  const int*   types     = (const int*)d_in[2];
  // d_in[3] = n_atoms_per_structure (constant 256, structures contiguous)
  const float* W0 = (const float*)d_in[4];
  const float* b0 = (const float*)d_in[5];
  const float* W1 = (const float*)d_in[6];
  const float* b1 = (const float*)d_in[7];
  const float* W2 = (const float*)d_in[8];
  const float* sbias = (const float*)d_in[9];

  float* e_atom  = (float*)d_out;
  float* e_total = e_atom + N_ATOMS;

  // workspace layout: 64 KB wpack | 64 KB cnt | 64 KB blockOff | 4 MB perm
  char* ws = (char*)d_ws;
  _Float16* wpack    = (_Float16*)ws;
  unsigned* cnt      = (unsigned*)(ws + (64 << 10));
  unsigned* blockOff = (unsigned*)(ws + (128 << 10));
  int*      perm     = (int*)(ws + (192 << 10));

  const size_t lds_bytes = WLDS_HALVES * 2 + 8 * STAGE_HALVES * 2;   // 80 KB

  // deterministic stable counting sort by type
  nep_hist<<<N_BLK, 256, 0, stream>>>(types, cnt);
  nep_offsets<<<1, 256, 0, stream>>>(cnt, blockOff);
  nep_scatter<<<N_BLK, 256, 0, stream>>>(types, blockOff, perm);
  // repack weights into WMMA B-fragment order
  nep_repack_w<<<128, 256, 0, stream>>>(W0, W1, wpack);
  // per-atom MLP over sorted windows: ~1 type pass per wave, weights in LDS
  nep_mlp_wmma<<<N_PAIRS / 8, 256, lds_bytes, stream>>>(
      g_radial, g_angular, types, perm, wpack, b0, b1, W2, sbias, e_atom);
  // deterministic segment sum over structures
  nep_struct_sum<<<N_STRUCT, 256, 0, stream>>>(e_atom, e_total);
}
